// GroupQueryAttention_91293824844143
// MI455X (gfx1250) — matrix operand order repaired and checked by reading, not compiled, not run on hardware
//
#include <hip/hip_runtime.h>
#include <math.h>

// ---------------------------------------------------------------------------
// GQA forward for MI455X (gfx1250, wave32, WMMA).
// All GEMMs + attention matmuls use v_wmma_f32_16x16x32_bf16.
// Wide (b128) global access everywhere; K tiles staged via async-to-LDS.
// ---------------------------------------------------------------------------

constexpr int S_LEN = 2048;     // sequence length
constexpr int HID   = 4096;     // hidden size
constexpr int NHEAD = 32;       // query heads
constexpr int NKVH  = 8;        // kv heads
constexpr int HDIM  = 128;      // head dim
constexpr int GRP   = NHEAD / NKVH;   // 4
constexpr int KVWID = NKVH * HDIM;    // 1024

typedef __bf16 bf16_t;
typedef __attribute__((ext_vector_type(16))) __bf16 v16bf;
typedef __attribute__((ext_vector_type(8)))  __bf16 v8bf;
typedef __attribute__((ext_vector_type(8)))  float  v8f;
typedef __attribute__((ext_vector_type(4)))  int    v4i;

#define AS_GLOBAL __attribute__((address_space(1)))
#define AS_LDS    __attribute__((address_space(3)))

#if defined(__has_builtin)
#if __has_builtin(__builtin_amdgcn_global_load_async_to_lds_b128)
#define HAVE_ASYNC_LDS 1
#endif
#endif
#ifndef HAVE_ASYNC_LDS
#define HAVE_ASYNC_LDS 0
#endif

union BF16x16 { v16bf v; v8bf h[2]; };

static __device__ __forceinline__ bf16_t f2bf(float f) {
  union { float f; unsigned u; } x; x.f = f;
  unsigned r = x.u + 0x7FFFu + ((x.u >> 16) & 1u);   // round-to-nearest-even
  union { unsigned short s; bf16_t b; } y; y.s = (unsigned short)(r >> 16);
  return y.b;
}

static __device__ __forceinline__ unsigned pack2(bf16_t a, bf16_t b) {
  union { bf16_t b; unsigned short s; } x, y; x.b = a; y.b = b;
  return (unsigned)x.s | ((unsigned)y.s << 16);
}

// Assemble a v16bf WMMA operand from two 16-byte-aligned 8-element chunks.
static __device__ __forceinline__ v16bf ld_pair(const bf16_t* p0, const bf16_t* p1) {
  BF16x16 u;
  u.h[0] = *(const v8bf*)p0;
  u.h[1] = *(const v8bf*)p1;
  return u.v;
}

static __device__ __forceinline__ v8f wmma_bf16(v16bf a, v16bf b, v8f c) {
  // (neg_a, A, neg_b, B, c_mod, C, reuse_a, reuse_b)
  return __builtin_amdgcn_wmma_f32_16x16x32_bf16(false, a, false, b, (short)0, c,
                                                 false, false);
}

// ---------------------------------------------------------------------------
// fp32 -> bf16, 8 elements per thread (float4 x2 in, v8bf out).  n8 = n/8.
// ---------------------------------------------------------------------------
__global__ void f32_to_bf16v(const float* __restrict__ src, bf16_t* __restrict__ dst,
                             long n8) {
  long stride = (long)gridDim.x * blockDim.x;
  for (long c = (long)blockIdx.x * blockDim.x + threadIdx.x; c < n8; c += stride) {
    const float* s = src + c * 8;
    float4 a = *(const float4*)s;
    float4 b = *(const float4*)(s + 4);
    v8bf o;
    o[0] = f2bf(a.x); o[1] = f2bf(a.y); o[2] = f2bf(a.z); o[3] = f2bf(a.w);
    o[4] = f2bf(b.x); o[5] = f2bf(b.y); o[6] = f2bf(b.z); o[7] = f2bf(b.w);
    *(v8bf*)(dst + c * 8) = o;
  }
}

// ---------------------------------------------------------------------------
// bf16 GEMM: C[M,N] (fp32) = A[M,K] @ B[K,N], both row-major bf16.
// Block: 128 threads = 4 waves. Tile 64(M) x 128(N); wave w owns rows w*16..+15.
// B K-slab transposed into LDS ([n][k], stride 40) with b128 global loads and
// packed b32 LDS stores; 8 WMMA per wave per K=32 chunk.
// ---------------------------------------------------------------------------
__global__ __launch_bounds__(128) void gemm_bf16(const bf16_t* __restrict__ A,
                                                 const bf16_t* __restrict__ B,
                                                 float* __restrict__ C,
                                                 int M, int N, int K) {
  __shared__ bf16_t Bt[128 * 40];   // [n][k] stride 40 (80B = 5x16B, banks spread)
  const int tid  = threadIdx.x;
  const int wave = tid >> 5;
  const int lane = tid & 31;
  const int m0   = blockIdx.y * 64 + wave * 16;
  const int n0   = blockIdx.x * 128;
  const int lcol = lane & 15;
  const int hi   = (lane & 16) ? 1 : 0;

  v8f acc[8];
#pragma unroll
  for (int j = 0; j < 8; ++j) acc[j] = (v8f)0.0f;

  for (int k0 = 0; k0 < K; k0 += 32) {
    __syncthreads();
    // B[k0..k0+31][n0..n0+127]: 16 k-pairs x 16 n-groups = 256 items, 2/thread.
    // Coalesced v8bf loads; transpose via packed b32 LDS stores.
    for (int idx = tid; idx < 256; idx += 128) {
      int kp = idx >> 4;        // k pair 0..15
      int g  = idx & 15;        // n group 0..15
      const bf16_t* bsrc = &B[(size_t)(k0 + 2 * kp) * N + n0 + 8 * g];
      v8bf r0 = *(const v8bf*)bsrc;
      v8bf r1 = *(const v8bf*)(bsrc + N);
#pragma unroll
      for (int e = 0; e < 8; ++e)
        *(unsigned*)&Bt[(8 * g + e) * 40 + 2 * kp] = pack2(r0[e], r1[e]);
    }
    __syncthreads();

    // A fragment: lane<16 -> row m0+lane, k={0..7,16..23}; lane>=16 -> {8..15,24..31}
    const bf16_t* ap = A + (size_t)(m0 + lcol) * K + k0 + hi * 8;
    v16bf af = ld_pair(ap, ap + 16);

#pragma unroll
    for (int j = 0; j < 8; ++j) {
      // B fragment: lane<16 -> col 16j+lane, k={0..7,16..23}; lane>=16 -> {8..15,24..31}
      const bf16_t* bp = &Bt[(16 * j + lcol) * 40 + hi * 8];
      v16bf bf = ld_pair(bp, bp + 16);
      acc[j] = wmma_bf16(af, bf, acc[j]);
    }
  }

  // C layout: VGPR r, lanes 0-15 -> (row m0+r, col), lanes 16-31 -> (row m0+r+8, col)
#pragma unroll
  for (int j = 0; j < 8; ++j)
#pragma unroll
    for (int r = 0; r < 8; ++r) {
      int row = m0 + r + hi * 8;
      int col = n0 + 16 * j + lcol;
      C[(size_t)row * N + col] = acc[j][r];
    }
}

// ---------------------------------------------------------------------------
// RoPE (base 500000) on Q/K + bf16 conversion of Q/K/V, 8 elements per thread.
// position_ids are arange(s) -> t = row index. 8 | 64, so a chunk of 8 head-dim
// elements never straddles the rotate_half boundary.
// ---------------------------------------------------------------------------
__global__ void rope_convert(const float* __restrict__ Q32,
                             const float* __restrict__ K32,
                             const float* __restrict__ V32,
                             bf16_t* __restrict__ Qb,
                             bf16_t* __restrict__ Kb,
                             bf16_t* __restrict__ Vb) {
  const long stride = (long)gridDim.x * blockDim.x;
  const long base   = (long)blockIdx.x * blockDim.x + threadIdx.x;
  const float lnb   = 13.122363377404328f;   // ln(500000)

  const long nq8 = (long)S_LEN * HID / 8;
  for (long ch = base; ch < nq8; ch += stride) {
    long b0 = ch * 8;
    int t = (int)(b0 / HID);
    int d = (int)(b0 % HID) & (HDIM - 1);    // multiple of 8
    const float* xp = Q32 + b0;
    const float* pp = xp + ((d < 64) ? 64 : -64);
    float4 x0 = *(const float4*)xp, x1 = *(const float4*)(xp + 4);
    float4 p0 = *(const float4*)pp, p1 = *(const float4*)(pp + 4);
    float xs[8] = {x0.x, x0.y, x0.z, x0.w, x1.x, x1.y, x1.z, x1.w};
    float ps[8] = {p0.x, p0.y, p0.z, p0.w, p1.x, p1.y, p1.z, p1.w};
    v8bf o;
#pragma unroll
    for (int e = 0; e < 8; ++e) {
      float inv = __expf(-((float)((d & 63) + e) / 64.0f) * lnb);
      float ang = (float)t * inv;
      float rh  = (d < 64) ? -ps[e] : ps[e];
      o[e] = f2bf(xs[e] * cosf(ang) + rh * sinf(ang));
    }
    *(v8bf*)(Qb + b0) = o;
  }

  const long nk8 = (long)S_LEN * KVWID / 8;
  for (long ch = base; ch < nk8; ch += stride) {
    long b0 = ch * 8;
    int t = (int)(b0 / KVWID);
    int d = (int)(b0 % KVWID) & (HDIM - 1);
    const float* xp = K32 + b0;
    const float* pp = xp + ((d < 64) ? 64 : -64);
    float4 x0 = *(const float4*)xp, x1 = *(const float4*)(xp + 4);
    float4 p0 = *(const float4*)pp, p1 = *(const float4*)(pp + 4);
    float xs[8] = {x0.x, x0.y, x0.z, x0.w, x1.x, x1.y, x1.z, x1.w};
    float ps[8] = {p0.x, p0.y, p0.z, p0.w, p1.x, p1.y, p1.z, p1.w};
    v8bf o;
#pragma unroll
    for (int e = 0; e < 8; ++e) {
      float inv = __expf(-((float)((d & 63) + e) / 64.0f) * lnb);
      float ang = (float)t * inv;
      float rh  = (d < 64) ? -ps[e] : ps[e];
      o[e] = f2bf(xs[e] * cosf(ang) + rh * sinf(ang));
    }
    *(v8bf*)(Kb + b0) = o;
  }

  for (long ch = base; ch < nk8; ch += stride) {
    long b0 = ch * 8;
    float4 a = *(const float4*)(V32 + b0);
    float4 b = *(const float4*)(V32 + b0 + 4);
    v8bf o;
    o[0] = f2bf(a.x); o[1] = f2bf(a.y); o[2] = f2bf(a.z); o[3] = f2bf(a.w);
    o[4] = f2bf(b.x); o[5] = f2bf(b.y); o[6] = f2bf(b.z); o[7] = f2bf(b.w);
    *(v8bf*)(Vb + b0) = o;
  }
}

// ---------------------------------------------------------------------------
// Flash-style GQA attention. Grid: (S/64, NH). Block: 128 threads = 4 waves,
// wave w owns 16 query rows. K/V 64-key tiles staged in LDS (K via async-to-LDS
// when the toolchain exposes it), online softmax in fp32, P bounced through
// per-wave LDS into WMMA A layout.
// ---------------------------------------------------------------------------
__global__ __launch_bounds__(128) void gqa_attn(const bf16_t* __restrict__ Qb,
                                                const bf16_t* __restrict__ Kb,
                                                const bf16_t* __restrict__ Vb,
                                                bf16_t* __restrict__ Cx) {
  __shared__ bf16_t Kt[64 * 136];     // [key][d]  stride 136 (272B = 17x16B)
  __shared__ bf16_t Vt[128 * 72];     // [d][key]  stride 72  (144B = 9x16B)
  __shared__ bf16_t Pt[4][16 * 72];   // per-wave P staging, stride 72

  const int tid  = threadIdx.x;
  const int wave = tid >> 5;
  const int lane = tid & 31;
  const int h    = blockIdx.y;
  const int kvh  = h / GRP;
  const int q0   = blockIdx.x * 64;
  const int qrow = q0 + wave * 16;
  const int lcol = lane & 15;
  const int hi   = (lane & 16) ? 1 : 0;
  const int half = hi * 8;            // C/D layout row offset for upper lanes

  // Q A-fragments, 4 chunks of K=32 along HD, loaded once from global
  v16bf qa[4];
#pragma unroll
  for (int c = 0; c < 4; ++c) {
    const bf16_t* qp = Qb + (size_t)(qrow + lcol) * HID + h * HDIM + 32 * c + hi * 8;
    qa[c] = ld_pair(qp, qp + 16);
  }

  v8f ctx[8];
  float m_i[8], l_i[8];
#pragma unroll
  for (int t = 0; t < 8; ++t) ctx[t] = (v8f)0.0f;
#pragma unroll
  for (int r = 0; r < 8; ++r) { m_i[r] = -3.0e38f; l_i[r] = 0.0f; }

  const float sc  = 0.088388347648318447f;   // 1/sqrt(128)
  const int  nkb  = blockIdx.x + 1;          // causal: key blocks 0..bx

  for (int kb = 0; kb < nkb; ++kb) {
    const int kbase = kb * 64;
    __syncthreads();
    // K/V tile: 32 key-pairs x 16 d-groups = 512 items, 4 per thread.
    // v8bf coalesced global loads; Kt rows stored b128; Vt transposed via b32.
    for (int i = tid; i < 512; i += 128) {
      int kp = i >> 4;        // key pair
      int g  = i & 15;        // d group
      size_t g0 = (size_t)(kbase + 2 * kp) * KVWID + kvh * HDIM + 8 * g;
#if HAVE_ASYNC_LDS
      __builtin_amdgcn_global_load_async_to_lds_b128(
          (AS_GLOBAL v4i*)&Kb[g0],
          (AS_LDS v4i*)&Kt[(2 * kp) * 136 + 8 * g], 0, 0);
      __builtin_amdgcn_global_load_async_to_lds_b128(
          (AS_GLOBAL v4i*)&Kb[g0 + KVWID],
          (AS_LDS v4i*)&Kt[(2 * kp + 1) * 136 + 8 * g], 0, 0);
#else
      *(v8bf*)&Kt[(2 * kp) * 136 + 8 * g]     = *(const v8bf*)&Kb[g0];
      *(v8bf*)&Kt[(2 * kp + 1) * 136 + 8 * g] = *(const v8bf*)&Kb[g0 + KVWID];
#endif
      v8bf v0 = *(const v8bf*)&Vb[g0];
      v8bf v1 = *(const v8bf*)&Vb[g0 + KVWID];
#pragma unroll
      for (int e = 0; e < 8; ++e)
        *(unsigned*)&Vt[(8 * g + e) * 72 + 2 * kp] = pack2(v0[e], v1[e]);
    }
#if HAVE_ASYNC_LDS
    asm volatile("s_wait_asynccnt 0x0" ::: "memory");
#endif
    if (kb + 1 < nkb)
      __builtin_prefetch(&Kb[(size_t)(kbase + 64 + (tid >> 1)) * KVWID + kvh * HDIM],
                         0, 1);
    __syncthreads();

    // scores S = Q @ K^T : four 16x16 tiles along keys, K-loop over HD (4x32)
    v8f st[4];
#pragma unroll
    for (int j = 0; j < 4; ++j) {
      v8f s = (v8f)0.0f;
#pragma unroll
      for (int c = 0; c < 4; ++c) {
        const bf16_t* kp2 = &Kt[(16 * j + lcol) * 136 + 32 * c + hi * 8];
        v16bf bfr = ld_pair(kp2, kp2 + 16);
        s = wmma_bf16(qa[c], bfr, s);
      }
      st[j] = s;
    }

    // online softmax (fp32) with causal mask
#pragma unroll
    for (int r = 0; r < 8; ++r) {
      const int qg = qrow + r + half;
      float vv[4];
#pragma unroll
      for (int j = 0; j < 4; ++j) {
        int kg = kbase + 16 * j + lcol;
        float v = st[j][r] * sc;
        vv[j] = (kg > qg) ? -3.0e38f : v;
      }
      float mloc = fmaxf(fmaxf(vv[0], vv[1]), fmaxf(vv[2], vv[3]));
#pragma unroll
      for (int o = 8; o >= 1; o >>= 1)
        mloc = fmaxf(mloc, __shfl_xor(mloc, o, 32));
      float mnew  = fmaxf(m_i[r], mloc);
      float alpha = __expf(m_i[r] - mnew);
      float rs = 0.0f;
#pragma unroll
      for (int j = 0; j < 4; ++j) {
        float p = __expf(vv[j] - mnew);
        rs += p;
        st[j][r] = p;
      }
#pragma unroll
      for (int o = 8; o >= 1; o >>= 1)
        rs += __shfl_xor(rs, o, 32);
      l_i[r] = l_i[r] * alpha + rs;
      m_i[r] = mnew;
#pragma unroll
      for (int t = 0; t < 8; ++t) ctx[t][r] *= alpha;
    }

    // stage P (bf16) to per-wave LDS patch in row-major [m][key] form
#pragma unroll
    for (int j = 0; j < 4; ++j)
#pragma unroll
      for (int r = 0; r < 8; ++r)
        Pt[wave][(r + half) * 72 + 16 * j + lcol] = f2bf(st[j][r]);

    // ctx += P @ V : two K=32 chunks over the 64 keys, 8 d-tiles of 16
#pragma unroll
    for (int c = 0; c < 2; ++c) {
      const bf16_t* pp = &Pt[wave][lcol * 72 + 32 * c + hi * 8];
      v16bf pa = ld_pair(pp, pp + 16);
#pragma unroll
      for (int t = 0; t < 8; ++t) {
        const bf16_t* vp = &Vt[(16 * t + lcol) * 72 + 32 * c + hi * 8];
        v16bf vb = ld_pair(vp, vp + 16);
        ctx[t] = wmma_bf16(pa, vb, ctx[t]);
      }
    }
  }

  // normalize + store ctx (bf16) in [s][NH*HD] layout for the output GEMM
#pragma unroll
  for (int t = 0; t < 8; ++t)
#pragma unroll
    for (int r = 0; r < 8; ++r) {
      int row = qrow + r + half;
      int col = h * HDIM + 16 * t + lcol;
      Cx[(size_t)row * HID + col] = f2bf(ctx[t][r] / l_i[r]);
    }
}

// ---------------------------------------------------------------------------
// Host launch
// ---------------------------------------------------------------------------
extern "C" void kernel_launch(void* const* d_in, const int* in_sizes, int n_in,
                              void* d_out, int out_size, void* d_ws, size_t ws_size,
                              hipStream_t stream) {
  (void)in_sizes; (void)n_in; (void)out_size; (void)ws_size;
  const float* hs = (const float*)d_in[0];
  // d_in[1] position_ids: arange -> implicit; d_in[2] attention_mask: causal -> implicit
  const float* Wq = (const float*)d_in[3];
  const float* Wk = (const float*)d_in[4];
  const float* Wv = (const float*)d_in[5];
  const float* Wo = (const float*)d_in[6];
  float* out = (float*)d_out;

  char* ws = (char*)d_ws;
  size_t off = 0;
  auto take = [&](size_t bytes) -> char* {
    char* p = ws + off;
    off += (bytes + 255) & ~(size_t)255;
    return p;
  };
  bf16_t* Xb  = (bf16_t*)take((size_t)S_LEN * HID * 2);
  bf16_t* Wqb = (bf16_t*)take((size_t)HID * HID * 2);
  bf16_t* Wkb = (bf16_t*)take((size_t)HID * KVWID * 2);
  bf16_t* Wvb = (bf16_t*)take((size_t)HID * KVWID * 2);
  bf16_t* Wob = (bf16_t*)take((size_t)HID * HID * 2);
  float*  Q32 = (float*)take((size_t)S_LEN * HID * 4);
  float*  K32 = (float*)take((size_t)S_LEN * KVWID * 4);
  float*  V32 = (float*)take((size_t)S_LEN * KVWID * 4);
  bf16_t* Qb  = (bf16_t*)take((size_t)S_LEN * HID * 2);
  bf16_t* Kb  = (bf16_t*)take((size_t)S_LEN * KVWID * 2);
  bf16_t* Vb  = (bf16_t*)take((size_t)S_LEN * KVWID * 2);
  bf16_t* Cxb = (bf16_t*)take((size_t)S_LEN * HID * 2);

  // 1) fp32 -> bf16 conversions (8-wide)
  f32_to_bf16v<<<1024, 256, 0, stream>>>(hs, Xb, (long)S_LEN * HID / 8);
  f32_to_bf16v<<<2048, 256, 0, stream>>>(Wq, Wqb, (long)HID * HID / 8);
  f32_to_bf16v<<<512,  256, 0, stream>>>(Wk, Wkb, (long)HID * KVWID / 8);
  f32_to_bf16v<<<512,  256, 0, stream>>>(Wv, Wvb, (long)HID * KVWID / 8);
  f32_to_bf16v<<<2048, 256, 0, stream>>>(Wo, Wob, (long)HID * HID / 8);

  // 2) QKV projections (WMMA bf16, fp32 out)
  gemm_bf16<<<dim3(HID / 128,   S_LEN / 64), 128, 0, stream>>>(Xb, Wqb, Q32, S_LEN, HID,   HID);
  gemm_bf16<<<dim3(KVWID / 128, S_LEN / 64), 128, 0, stream>>>(Xb, Wkb, K32, S_LEN, KVWID, HID);
  gemm_bf16<<<dim3(KVWID / 128, S_LEN / 64), 128, 0, stream>>>(Xb, Wvb, V32, S_LEN, KVWID, HID);

  // 3) RoPE + bf16 conversion (8-wide)
  rope_convert<<<2048, 256, 0, stream>>>(Q32, K32, V32, Qb, Kb, Vb);

  // 4) flash-style GQA attention (WMMA bf16)
  gqa_attn<<<dim3(S_LEN / 64, NHEAD), 128, 0, stream>>>(Qb, Kb, Vb, Cxb);

  // 5) output projection -> fp32 d_out
  gemm_bf16<<<dim3(HID / 128, S_LEN / 64), 128, 0, stream>>>(Cxb, Wob, out, S_LEN, HID, HID);
}
